// AttentionTSSA_52888227283202
// MI455X (gfx1250) — matrix operand order repaired and checked by reading, not compiled, not run on hardware
//
#include <hip/hip_runtime.h>
#include <hip/hip_bf16.h>

// ---------------------------------------------------------------------------
// AttentionTSSA for MI455X (gfx1250): bf16 WMMA GEMMs + f32 reductions.
// B=8, N=4096, D=1024, HEADS=16, d=64.
// GEMM: software-pipelined (double-buffered fragments), wave tile 32x64.
// ---------------------------------------------------------------------------

typedef __attribute__((ext_vector_type(16))) __bf16 v16bf;
typedef __attribute__((ext_vector_type(8)))  float  v8f;

static constexpr int Bb   = 8;
static constexpr int Nn   = 4096;
static constexpr int Dd   = 1024;
static constexpr int Hh   = 16;
static constexpr int dH   = 64;
static constexpr int Mrows = Bb * Nn;              // 32768

union FragAB { v16bf v; uint4 q[2]; };
union Pack4  { __bf16 h[4]; uint2 u; };

__device__ __forceinline__ FragAB ldfrag(const __bf16* __restrict__ p, int off, int gap)
{
    FragAB f;
    f.q[0] = *reinterpret_cast<const uint4*>(p + off);
    f.q[1] = *reinterpret_cast<const uint4*>(p + off + gap);
    return f;
}

// ---------------------------------------------------------------------------
// f32 -> bf16 conversion (vectorized x4)
// ---------------------------------------------------------------------------
__global__ __launch_bounds__(256)
void f32_to_bf16_kernel(const float* __restrict__ in, __bf16* __restrict__ out, int n4)
{
    int i = blockIdx.x * blockDim.x + threadIdx.x;
    if (i >= n4) return;
    float4 f = reinterpret_cast<const float4*>(in)[i];
    Pack4 p;
    p.h[0] = (__bf16)f.x; p.h[1] = (__bf16)f.y;
    p.h[2] = (__bf16)f.z; p.h[3] = (__bf16)f.w;
    reinterpret_cast<uint2*>(out)[i] = p.u;
}

// ---------------------------------------------------------------------------
// GEMM: C[M,1024] = A[M,1024](bf16) * W[1024,1024](bf16 row-major [col,k])^T
//   mode 0: scatter-store f32 into w layout [b, h, n, d]   (for GEMM1)
//   mode 1: C[m,col] = acc + bias[col], row-major f32      (for GEMM2)
// Block: 256 threads = 8 waves. Block tile 256(M) x 64(N).
// Wave tile: 32(M) x 64(N) = 8 accumulators (2 M-tiles x 4 N-tiles).
// K loop is double-buffered: loads for k+1 are issued before the 8 WMMAs
// of step k so v_wmma execution overlaps next-tile global loads.
// ---------------------------------------------------------------------------
__global__ __launch_bounds__(256)
void gemm_bf16_wmma(const __bf16* __restrict__ A, const __bf16* __restrict__ W,
                    float* __restrict__ C, const float* __restrict__ bias, int mode)
{
    constexpr int K = Dd;
    const int lane = threadIdx.x & 31;
    const int wv   = threadIdx.x >> 5;
    const int half = lane >> 4;             // A: K 0-15 vs 16-31
    const int mlo  = lane & 15;
    const int mwave = blockIdx.x * 256 + wv * 32;
    const int n0    = blockIdx.y * 64;

    // A fragment pointers (two 16-row tiles), pre-offset by half*16
    const __bf16* __restrict__ a0 = A + (size_t)(mwave + mlo) * K + half * 16;
    const __bf16* __restrict__ a1 = a0 + (size_t)16 * K;
    // B fragment pointers (4 column tiles), pre-offset by half*16
    const __bf16* __restrict__ w0 = W + (size_t)(n0 +  0 + mlo) * K + half * 16;
    const __bf16* __restrict__ w1 = W + (size_t)(n0 + 16 + mlo) * K + half * 16;
    const __bf16* __restrict__ w2 = W + (size_t)(n0 + 32 + mlo) * K + half * 16;
    const __bf16* __restrict__ w3 = W + (size_t)(n0 + 48 + mlo) * K + half * 16;

    v8f acc[2][4];
#pragma unroll
    for (int i = 0; i < 2; ++i)
#pragma unroll
        for (int j = 0; j < 4; ++j) acc[i][j] = (v8f){};

    // ---- prologue: load k0 = 0 fragments ----
    FragAB ca0 = ldfrag(a0, 0, 8);
    FragAB ca1 = ldfrag(a1, 0, 8);
    FragAB cb0 = ldfrag(w0, 0, 8);
    FragAB cb1 = ldfrag(w1, 0, 8);
    FragAB cb2 = ldfrag(w2, 0, 8);
    FragAB cb3 = ldfrag(w3, 0, 8);

#pragma unroll 2
    for (int k0 = 32; k0 < K; k0 += 32) {
        // issue next-step loads first (overlaps with WMMA execution below)
        FragAB na0 = ldfrag(a0, k0, 8);
        FragAB na1 = ldfrag(a1, k0, 8);
        FragAB nb0 = ldfrag(w0, k0, 8);
        FragAB nb1 = ldfrag(w1, k0, 8);
        FragAB nb2 = ldfrag(w2, k0, 8);
        FragAB nb3 = ldfrag(w3, k0, 8);

        acc[0][0] = __builtin_amdgcn_wmma_f32_16x16x32_bf16(false, ca0.v, false, cb0.v, (short)0, acc[0][0], false, false);
        acc[0][1] = __builtin_amdgcn_wmma_f32_16x16x32_bf16(false, ca0.v, false, cb1.v, (short)0, acc[0][1], false, false);
        acc[0][2] = __builtin_amdgcn_wmma_f32_16x16x32_bf16(false, ca0.v, false, cb2.v, (short)0, acc[0][2], false, false);
        acc[0][3] = __builtin_amdgcn_wmma_f32_16x16x32_bf16(false, ca0.v, false, cb3.v, (short)0, acc[0][3], false, false);
        acc[1][0] = __builtin_amdgcn_wmma_f32_16x16x32_bf16(false, ca1.v, false, cb0.v, (short)0, acc[1][0], false, false);
        acc[1][1] = __builtin_amdgcn_wmma_f32_16x16x32_bf16(false, ca1.v, false, cb1.v, (short)0, acc[1][1], false, false);
        acc[1][2] = __builtin_amdgcn_wmma_f32_16x16x32_bf16(false, ca1.v, false, cb2.v, (short)0, acc[1][2], false, false);
        acc[1][3] = __builtin_amdgcn_wmma_f32_16x16x32_bf16(false, ca1.v, false, cb3.v, (short)0, acc[1][3], false, false);

        ca0 = na0; ca1 = na1;
        cb0 = nb0; cb1 = nb1; cb2 = nb2; cb3 = nb3;
    }

    // ---- epilogue WMMAs for the last K step ----
    acc[0][0] = __builtin_amdgcn_wmma_f32_16x16x32_bf16(false, ca0.v, false, cb0.v, (short)0, acc[0][0], false, false);
    acc[0][1] = __builtin_amdgcn_wmma_f32_16x16x32_bf16(false, ca0.v, false, cb1.v, (short)0, acc[0][1], false, false);
    acc[0][2] = __builtin_amdgcn_wmma_f32_16x16x32_bf16(false, ca0.v, false, cb2.v, (short)0, acc[0][2], false, false);
    acc[0][3] = __builtin_amdgcn_wmma_f32_16x16x32_bf16(false, ca0.v, false, cb3.v, (short)0, acc[0][3], false, false);
    acc[1][0] = __builtin_amdgcn_wmma_f32_16x16x32_bf16(false, ca1.v, false, cb0.v, (short)0, acc[1][0], false, false);
    acc[1][1] = __builtin_amdgcn_wmma_f32_16x16x32_bf16(false, ca1.v, false, cb1.v, (short)0, acc[1][1], false, false);
    acc[1][2] = __builtin_amdgcn_wmma_f32_16x16x32_bf16(false, ca1.v, false, cb2.v, (short)0, acc[1][2], false, false);
    acc[1][3] = __builtin_amdgcn_wmma_f32_16x16x32_bf16(false, ca1.v, false, cb3.v, (short)0, acc[1][3], false, false);

    // C/D layout: element e -> row = half*8 + e, col = lane&15 (within tile)
#pragma unroll
    for (int mt = 0; mt < 2; ++mt) {
        const int mbase = mwave + mt * 16 + half * 8;
#pragma unroll
        for (int t = 0; t < 4; ++t) {
            const int col = n0 + t * 16 + mlo;
#pragma unroll
            for (int e = 0; e < 8; ++e) {
                const int m = mbase + e;
                const float v = acc[mt][t][e];
                if (mode == 0) {
                    // m = b*4096 + n ; col = h*64 + d  ->  w[b][h][n][d]
                    const int b  = m >> 12, ns = m & 4095;
                    const int hh = col >> 6, dd = col & 63;
                    C[(((size_t)(b * Hh + hh) * Nn) + ns) * dH + dd] = v;
                } else {
                    C[(size_t)m * Dd + col] = v + bias[col];
                }
            }
        }
    }
}

// ---------------------------------------------------------------------------
// Per (b,h,d): 1/max(sum_n w^2, eps^2)   (== 1/max(norm,1e-12)^2)
// ---------------------------------------------------------------------------
__global__ __launch_bounds__(256)
void col_norms_kernel(const float* __restrict__ w, float* __restrict__ rnorm2)
{
    const int bh = blockIdx.x;                 // 0..127
    const int dd = threadIdx.x & 63;
    const int g  = threadIdx.x >> 6;           // 0..3
    const float* __restrict__ base = w + (size_t)bh * Nn * dH;
    float s = 0.f;
    for (int n = g; n < Nn; n += 4) {
        float v = base[(size_t)n * dH + dd];
        s = fmaf(v, v, s);
    }
    __shared__ float red[256];
    red[threadIdx.x] = s;
    __syncthreads();
    if (g == 0) {
        float t = red[dd] + red[64 + dd] + red[128 + dd] + red[192 + dd];
        rnorm2[bh * dH + dd] = 1.0f / fmaxf(t, 1e-24f);
    }
}

// ---------------------------------------------------------------------------
// Per (b,n): logits over 16 heads, softmax over heads -> Pi[b,h,n]
// ---------------------------------------------------------------------------
__global__ __launch_bounds__(256)
void softmax_heads_kernel(const float* __restrict__ w, const float* __restrict__ rnorm2,
                          const float* __restrict__ temp, float* __restrict__ Pi)
{
    const int b = blockIdx.x >> 12;
    const int n = blockIdx.x & 4095;
    const int h   = threadIdx.x >> 4;          // 0..15
    const int sub = threadIdx.x & 15;          // 0..15 -> 4 d's each
    const size_t wbase = ((size_t)(b * Hh + h) * Nn + n) * dH + sub * 4;
    const float4 wv = *reinterpret_cast<const float4*>(w + wbase);
    const float4 rv = *reinterpret_cast<const float4*>(rnorm2 + (b * Hh + h) * dH + sub * 4);
    float s = wv.x * wv.x * rv.x + wv.y * wv.y * rv.y +
              wv.z * wv.z * rv.z + wv.w * wv.w * rv.w;

    __shared__ float red[16][16];
    __shared__ float logits[16];
    red[h][sub] = s;
    __syncthreads();
    if (sub == 0) {
        float t = 0.f;
#pragma unroll
        for (int i = 0; i < 16; ++i) t += red[h][i];
        logits[h] = t * temp[h];
    }
    __syncthreads();
    if (threadIdx.x < 16) {
        float mx = -1e30f;
#pragma unroll
        for (int i = 0; i < 16; ++i) mx = fmaxf(mx, logits[i]);
        float denom = 0.f;
#pragma unroll
        for (int i = 0; i < 16; ++i) denom += __expf(logits[i] - mx);
        const float p = __expf(logits[threadIdx.x] - mx) / denom;
        Pi[(size_t)(b * Hh + threadIdx.x) * Nn + n] = p;
    }
}

// ---------------------------------------------------------------------------
// Per (b,h): S = sum_n Pi ; dots[d] = sum_n (Pi/S') * w^2 ; attn = 1/(1+dots)
// ---------------------------------------------------------------------------
__global__ __launch_bounds__(256)
void pi_dots_kernel(const float* __restrict__ w, const float* __restrict__ Pi,
                    float* __restrict__ attn)
{
    const int bh = blockIdx.x;
    const float* __restrict__ pirow = Pi + (size_t)bh * Nn;
    __shared__ float red[256];
    __shared__ float rS_sh;

    float s = 0.f;
    for (int n = threadIdx.x; n < Nn; n += 256) s += pirow[n];
    red[threadIdx.x] = s;
    __syncthreads();
    for (int off = 128; off > 0; off >>= 1) {
        if ((int)threadIdx.x < off) red[threadIdx.x] += red[threadIdx.x + off];
        __syncthreads();
    }
    if (threadIdx.x == 0) rS_sh = 1.0f / (red[0] + 1e-8f);
    __syncthreads();
    const float rS = rS_sh;

    const int dd = threadIdx.x & 63;
    const int g  = threadIdx.x >> 6;
    const float* __restrict__ base = w + (size_t)bh * Nn * dH;
    float acc = 0.f;
    for (int n = g; n < Nn; n += 4) {
        float v = base[(size_t)n * dH + dd];
        acc = fmaf(v * v, pirow[n], acc);
    }
    __syncthreads();
    red[threadIdx.x] = acc;
    __syncthreads();
    if (g == 0) {
        float t = red[dd] + red[64 + dd] + red[128 + dd] + red[192 + dd];
        attn[bh * dH + dd] = 1.0f / (1.0f + t * rS);
    }
}

// ---------------------------------------------------------------------------
// Elementwise: outb[b, n, h*64+d] = bf16( -(w * Pi) * attn )
// ---------------------------------------------------------------------------
__global__ __launch_bounds__(256)
void epilogue_bf16_kernel(const float* __restrict__ w, const float* __restrict__ Pi,
                          const float* __restrict__ attn, __bf16* __restrict__ outb)
{
    const size_t i4 = ((size_t)blockIdx.x * blockDim.x + threadIdx.x) * 4;
    constexpr size_t total = (size_t)Bb * Hh * Nn * dH;
    if (i4 >= total) return;
    const int dd = (int)(i4 & 63);
    const size_t r = i4 >> 6;
    const int n  = (int)(r & 4095);
    const int bh = (int)(r >> 12);
    const float p = Pi[(size_t)bh * Nn + n];
    const float4 wv = *reinterpret_cast<const float4*>(w + i4);
    const float4 a4 = *reinterpret_cast<const float4*>(attn + bh * dH + dd);
    const int h = bh & 15, b = bh >> 4;
    const size_t o = ((size_t)(b * Nn + n)) * Dd + h * dH + dd;
    Pack4 pk;
    pk.h[0] = (__bf16)(-(wv.x * p) * a4.x);
    pk.h[1] = (__bf16)(-(wv.y * p) * a4.y);
    pk.h[2] = (__bf16)(-(wv.z * p) * a4.z);
    pk.h[3] = (__bf16)(-(wv.w * p) * a4.w);
    *reinterpret_cast<uint2*>(outb + o) = pk.u;
}

// ---------------------------------------------------------------------------
// Launch
// ---------------------------------------------------------------------------
extern "C" void kernel_launch(void* const* d_in, const int* in_sizes, int n_in,
                              void* d_out, int out_size, void* d_ws, size_t ws_size,
                              hipStream_t stream)
{
    const float* x      = (const float*)d_in[0];   // [8,4096,1024]
    const float* W_qkv  = (const float*)d_in[1];   // [1024,1024]
    const float* temp   = (const float*)d_in[2];   // [16,1]
    const float* W_out  = (const float*)d_in[3];   // [1024,1024]
    const float* b_out  = (const float*)d_in[4];   // [1024]
    float* out = (float*)d_out;                    // [8,4096,1024]

    char* ws = (char*)d_ws;
    size_t off = 0;
    auto salloc = [&](size_t bytes) -> void* {
        off = (off + 255) & ~(size_t)255;
        void* p = ws + off;
        off += bytes;
        return p;
    };

    const size_t nX  = (size_t)Mrows * Dd;         // 33.55M
    const size_t nW  = (size_t)Dd * Dd;            // 1.05M
    __bf16* Wq_b  = (__bf16*)salloc(nW * 2);       //   2 MB
    __bf16* Wo_b  = (__bf16*)salloc(nW * 2);       //   2 MB
    __bf16* abuf  = (__bf16*)salloc(nX * 2);       //  64 MB (xb, then outb)
    float*  wbuf  = (float*) salloc(nX * 4);       // 128 MB  w[b,h,n,d]
    float*  Pi    = (float*) salloc((size_t)Bb * Hh * Nn * 4);   // 2 MB
    float*  rnorm2= (float*) salloc((size_t)Bb * Hh * dH * 4);   // 32 KB
    float*  attn  = (float*) salloc((size_t)Bb * Hh * dH * 4);   // 32 KB
    (void)ws_size; (void)in_sizes; (void)n_in; (void)out_size;

    // 1) convert inputs to bf16
    f32_to_bf16_kernel<<<(int)(nX / 4 + 255) / 256, 256, 0, stream>>>(x, abuf, (int)(nX / 4));
    f32_to_bf16_kernel<<<(int)(nW / 4 + 255) / 256, 256, 0, stream>>>(W_qkv, Wq_b, (int)(nW / 4));
    f32_to_bf16_kernel<<<(int)(nW / 4 + 255) / 256, 256, 0, stream>>>(W_out, Wo_b, (int)(nW / 4));

    // 2) GEMM1: w[b,h,n,d] = x @ W_qkv^T   (WMMA bf16, f32 accum)
    gemm_bf16_wmma<<<dim3(Mrows / 256, Dd / 64), 256, 0, stream>>>(abuf, Wq_b, wbuf, nullptr, 0);

    // 3) reductions
    col_norms_kernel<<<Bb * Hh, 256, 0, stream>>>(wbuf, rnorm2);
    softmax_heads_kernel<<<Bb * Nn, 256, 0, stream>>>(wbuf, rnorm2, temp, Pi);
    pi_dots_kernel<<<Bb * Hh, 256, 0, stream>>>(wbuf, Pi, attn);

    // 4) elementwise epilogue -> bf16 A operand for GEMM2 (reuses abuf)
    epilogue_bf16_kernel<<<(int)(nX / 4 + 255) / 256, 256, 0, stream>>>(wbuf, Pi, attn, abuf);

    // 5) GEMM2: out = act @ W_out^T + b_out
    gemm_bf16_wmma<<<dim3(Mrows / 256, Dd / 64), 256, 0, stream>>>(abuf, Wo_b, out, b_out, 1);
}
